// WidthPredictor_28870770164150
// MI455X (gfx1250) — matrix operand order repaired and checked by reading, not compiled, not run on hardware
//
#include <hip/hip_runtime.h>
#include <math.h>

typedef __attribute__((ext_vector_type(16))) _Float16 v16h;
typedef __attribute__((ext_vector_type(8)))  _Float16 v8h;
typedef __attribute__((ext_vector_type(8)))  float    v8f;

#define IN_CH 512
#define MID2  1024   // MAX_MID*2
#define MID   512    // MAX_MID
#define BATCH 32
#define HW    4096   // 64*64

// ------------------------------------------------------------------
// Kernel 1: global average pool. One block per (b,c) row of 4096 floats.
// Reads 256 MB total -> this kernel is the HBM-bound bulk (~11us @ 23.3TB/s).
// Stores means as f16 (WMMA A operand for layer 1).
// ------------------------------------------------------------------
__global__ void gap_kernel(const float* __restrict__ x,
                           _Float16* __restrict__ gapf16) {
    __shared__ float sdata[256];
    const int t = threadIdx.x;
    const float4* xp = (const float4*)x + (size_t)blockIdx.x * (HW / 4);
    float s = 0.f;
#pragma unroll
    for (int i = 0; i < 4; ++i) {
        float4 v = xp[t + 256 * i];       // coalesced b128 loads
        s += (v.x + v.y) + (v.z + v.w);
    }
    sdata[t] = s;
    __syncthreads();
    for (int off = 128; off > 0; off >>= 1) {
        if (t < off) sdata[t] += sdata[t + off];
        __syncthreads();
    }
    if (t == 0) gapf16[blockIdx.x] = (_Float16)(sdata[0] * (1.0f / HW));
}

// ------------------------------------------------------------------
// Kernel 2: layer 1, h = relu(gap @ w1^T + b1).  [32,512]x[512,1024]
// One 16x16 tile per wave32 via v_wmma_f32_16x16x32_f16.
// Fragment layouts per CDNA5 ISA 7.12.2:
//   A (16x32 f16): lane m (both halves) holds row m; lanes<16 K=0..7,16..23;
//                  lanes>=16 K=8..15,24..31  -> two b128 loads per lane.
//   B (32x16 f16): lane holds column n; lanes<16 K=0..7,16..23; lanes>=16 K=8..15,24..31 (same as A)
//                  -> two runs of 8 contiguous K values from a row of w1 (f32),
//                  loaded as 4x float4 and converted to f16 in-register.
//   D (16x16 f32): VGPR j -> rows j (lanes<16) / j+8 (lanes>=16), col = lane&15.
// ------------------------------------------------------------------
__global__ void mlp1_kernel(const _Float16* __restrict__ gapf16,
                            const float* __restrict__ w1,
                            const float* __restrict__ b1,
                            _Float16* __restrict__ hf16) {
    const int wave   = threadIdx.x >> 5;
    const int lane   = threadIdx.x & 31;
    const int tile   = blockIdx.x * (blockDim.x >> 5) + wave; // 0..127
    const int mt     = tile & 1;    // 2 M tiles
    const int nt     = tile >> 1;   // 64 N tiles
    const int lanelo = lane & 15;
    const int ishi   = lane >> 4;
    const int n      = nt * 16 + lanelo;

    v8f c = {};
    for (int kk = 0; kk < IN_CH; kk += 32) {
        // A fragment from gap f16 (row-major [32][512])
        const _Float16* arow = gapf16 + (mt * 16 + lanelo) * IN_CH + kk + ishi * 8;
        v8h a0 = *(const v8h*)(arow);
        v8h a1 = *(const v8h*)(arow + 16);
        v16h a;
#pragma unroll
        for (int i = 0; i < 8; ++i) { a[i] = a0[i]; a[i + 8] = a1[i]; }

        // B fragment: B[k][n] = w1[n][k]; w1 row-major f32 [1024][512]
        const float* bp = w1 + (size_t)n * IN_CH + kk + ishi * 8;
        v16h b;
#pragma unroll
        for (int i = 0; i < 16; i += 4) {
            float4 f = *(const float4*)(bp + i + (i & 8));
            b[i + 0] = (_Float16)f.x; b[i + 1] = (_Float16)f.y;
            b[i + 2] = (_Float16)f.z; b[i + 3] = (_Float16)f.w;
        }
        c = __builtin_amdgcn_wmma_f32_16x16x32_f16(false, a, false, b,
                                                   (short)0, c, false, false);
    }
    const float bias = b1[n];
#pragma unroll
    for (int j = 0; j < 8; ++j) {
        const int row = mt * 16 + j + ishi * 8;
        float v = c[j] + bias;
        v = v > 0.f ? v : 0.f;                      // ReLU
        hf16[row * MID2 + n] = (_Float16)v;
    }
}

// ------------------------------------------------------------------
// Kernel 3: layer 2, scores = sigmoid(h @ w2^T + b2). [32,1024]x[1024,512]
// ------------------------------------------------------------------
__global__ void mlp2_kernel(const _Float16* __restrict__ hf16,
                            const float* __restrict__ w2,
                            const float* __restrict__ b2,
                            float* __restrict__ scores) {
    const int wave   = threadIdx.x >> 5;
    const int lane   = threadIdx.x & 31;
    const int tile   = blockIdx.x * (blockDim.x >> 5) + wave; // 0..63
    const int mt     = tile & 1;    // 2 M tiles
    const int nt     = tile >> 1;   // 32 N tiles
    const int lanelo = lane & 15;
    const int ishi   = lane >> 4;
    const int n      = nt * 16 + lanelo;

    v8f c = {};
    for (int kk = 0; kk < MID2; kk += 32) {
        const _Float16* arow = hf16 + (mt * 16 + lanelo) * MID2 + kk + ishi * 8;
        v8h a0 = *(const v8h*)(arow);
        v8h a1 = *(const v8h*)(arow + 16);
        v16h a;
#pragma unroll
        for (int i = 0; i < 8; ++i) { a[i] = a0[i]; a[i + 8] = a1[i]; }

        const float* bp = w2 + (size_t)n * MID2 + kk + ishi * 8;
        v16h b;
#pragma unroll
        for (int i = 0; i < 16; i += 4) {
            float4 f = *(const float4*)(bp + i + (i & 8));
            b[i + 0] = (_Float16)f.x; b[i + 1] = (_Float16)f.y;
            b[i + 2] = (_Float16)f.z; b[i + 3] = (_Float16)f.w;
        }
        c = __builtin_amdgcn_wmma_f32_16x16x32_f16(false, a, false, b,
                                                   (short)0, c, false, false);
    }
    const float bias = b2[n];
#pragma unroll
    for (int j = 0; j < 8; ++j) {
        const int row = mt * 16 + j + ishi * 8;
        const float v = c[j] + bias;
        scores[row * MID + n] = 1.0f / (1.0f + expf(-v));   // sigmoid
    }
}

// ------------------------------------------------------------------
// Kernel 4: scale = clamp(mean(scores),.25,1); k = clamp(ceil(512*scale),128,512)
// ------------------------------------------------------------------
__global__ void kcalc_kernel(const float* __restrict__ scores,
                             int* __restrict__ kout,
                             float* __restrict__ kout_f) {
    __shared__ float sdata[256];
    const int t = threadIdx.x;
    float s = 0.f;
    for (int i = t; i < BATCH * MID; i += 256) s += scores[i];
    sdata[t] = s;
    __syncthreads();
    for (int off = 128; off > 0; off >>= 1) {
        if (t < off) sdata[t] += sdata[t + off];
        __syncthreads();
    }
    if (t == 0) {
        const float mean  = sdata[0] * (1.0f / (BATCH * MID));
        const float scale = fminf(fmaxf(mean, 0.25f), 1.0f);
        int k = (int)ceilf((float)MID * scale);
        k = max(MID / 4, min(MID, k));     // k_min = max(4, 512//4) = 128
        *kout = k;
        if (kout_f) *kout_f = (float)k;
    }
}

// ------------------------------------------------------------------
// Kernel 5: per-row top-k mask without sorting.
// mask[i] = 1  iff  #{j : s_j > s_i} < k   (== s_i >= k-th largest, tie-safe)
// One block per row; row cached in LDS; inner loop is an LDS broadcast.
// ------------------------------------------------------------------
__global__ void mask_kernel(const float* __restrict__ scores,
                            const int* __restrict__ kptr,
                            float* __restrict__ out) {
    __shared__ float row[MID];
    const int t = threadIdx.x;
    const int r = blockIdx.x;
    row[t]       = scores[r * MID + t];
    row[t + 256] = scores[r * MID + t + 256];
    __syncthreads();
    const int k = *kptr;
#pragma unroll 1
    for (int e = t; e < MID; e += 256) {
        const float v = row[e];
        int cnt = 0;
        for (int j = 0; j < MID; ++j) cnt += (row[j] > v) ? 1 : 0;
        out[r * MID + e] = (cnt < k) ? 1.0f : 0.0f;
    }
}

// ------------------------------------------------------------------
extern "C" void kernel_launch(void* const* d_in, const int* in_sizes, int n_in,
                              void* d_out, int out_size, void* d_ws, size_t ws_size,
                              hipStream_t stream) {
    const float* x  = (const float*)d_in[0];
    const float* w1 = (const float*)d_in[1];
    const float* b1 = (const float*)d_in[2];
    const float* w2 = (const float*)d_in[3];
    const float* b2 = (const float*)d_in[4];
    float* out = (float*)d_out;

    // workspace layout (all 256B-aligned offsets): ~164 KB total
    char* ws = (char*)d_ws;
    _Float16* gapf16 = (_Float16*)(ws);            // 32*512  f16 = 32 KB
    _Float16* hf16   = (_Float16*)(ws + 32768);    // 32*1024 f16 = 64 KB
    float*    scores = (float*)   (ws + 98304);    // 32*512  f32 = 64 KB
    int*      kint   = (int*)     (ws + 163840);   // 4 B

    // k goes after the 16384 mask floats if the harness allotted room for it
    float* kslot = (out_size > BATCH * MID) ? (out + BATCH * MID) : nullptr;

    gap_kernel  <<<BATCH * IN_CH, 256, 0, stream>>>(x, gapf16);
    mlp1_kernel <<<16, 256, 0, stream>>>(gapf16, w1, b1, hf16);   // 128 tiles
    mlp2_kernel <<< 8, 256, 0, stream>>>(hf16, w2, b2, scores);   //  64 tiles
    kcalc_kernel<<< 1, 256, 0, stream>>>(scores, kint, kslot);
    mask_kernel <<<BATCH, 256, 0, stream>>>(scores, kint, out);
}
